// WindowAttention_14173392076862
// MI455X (gfx1250) — matrix operand order repaired and checked by reading, not compiled, not run on hardware
//
#include <hip/hip_runtime.h>

// ---------------- CDNA5 (gfx1250) fused window-attention ----------------
// wave32, WMMA bf16 16x16x32, f32 accumulate.
// One workgroup (256 thr = 8 waves) per 8x8 window; one wave per head.
// GEMM phases: each wave owns one 16-row tile and a contiguous column range,
// preloading all A fragments once and streaming B fragments from L2.

typedef __attribute__((ext_vector_type(16))) __bf16 v16bf;
typedef __attribute__((ext_vector_type(8)))  __bf16 v8bf;
typedef __attribute__((ext_vector_type(8)))  float  v8f;

#define DIM   384
#define NH    8
#define HD    48      // head dim
#define WS    8
#define TOKS  64      // WS*WS tokens per window
#define QKVW  1152    // 3*DIM
#define KSTEPS (DIM / 32)   // 12

// LDS layout (bytes):
//   [0      , 98304 )  qk   : 64 x 768 bf16 (Q scaled | K)   / later: 64x384 f32 proj out
//   [98304  , 147456)  vT   : 384 x 64 bf16 (V transposed)
//   [147456 , 196608)  xa   : 64 x 384 bf16 (input window, later attn output)
//   [196608 , 262144)  P    : 8 heads x 64 x 64 bf16 (softmax probs)
#define LDS_BYTES 262144

__device__ __forceinline__ v8bf zero8() { v8bf z = {}; return z; }

// A-fragment (16x32 bf16, M=lane&15): elems e -> k = koff+2*kbase+e,
// kbase = 8*(lane>=16).
__device__ __forceinline__ v16bf ld_fragA(const __bf16* row, int kbase, int koff) {
  union { v16bf v; v8bf h[2]; } u;
  u.h[0] = *(const v8bf*)(row + koff + 2 * kbase);
  u.h[1] = *(const v8bf*)(row + koff + 2 * kbase + 8);
  return u.v;
}
// A-fragment with upper k-group masked to zero (for head_dim 48 pad->64).
__device__ __forceinline__ v16bf ld_fragA_lo(const __bf16* row, int kbase, int koff) {
  union { v16bf v; v8bf h[2]; } u;
  u.h[0] = kbase ? zero8() : *(const v8bf*)(row + koff);
  u.h[1] = kbase ? zero8() : *(const v8bf*)(row + koff + 8);
  return u.v;
}
// B-fragment (32x16 bf16, N=lane&15): elems e -> k = koff + 16*(lane>=16) + e.
__device__ __forceinline__ v16bf ld_fragB(const __bf16* col, int lane, int koff) {
  int k0 = koff + ((lane >> 4) << 4);
  union { v16bf v; v8bf h[2]; } u;
  u.h[0] = *(const v8bf*)(col + k0);
  u.h[1] = *(const v8bf*)(col + k0 + 8);
  return u.v;
}

__global__ void convert_weights_kernel(const float* __restrict__ Wqkv,
                                       const float* __restrict__ Wproj,
                                       __bf16* __restrict__ WqkvT,
                                       __bf16* __restrict__ WprojT) {
  int idx = blockIdx.x * blockDim.x + threadIdx.x;
  const int NQ = QKVW * DIM;
  if (idx < NQ) {
    int n = idx / DIM, k = idx - n * DIM;
    WqkvT[idx] = (__bf16)Wqkv[k * QKVW + n];         // [n][k]
  } else if (idx < NQ + DIM * DIM) {
    int j = idx - NQ;
    int n = j / DIM, k = j - n * DIM;
    WprojT[j] = (__bf16)Wproj[k * DIM + n];          // [n][k]
  }
}

__global__ void __launch_bounds__(256)
window_attention_kernel(const float* __restrict__ x,
                        const float* __restrict__ b_qkv,
                        const float* __restrict__ b_proj,
                        const __bf16* __restrict__ WqkvT,
                        const __bf16* __restrict__ WprojT,
                        float* __restrict__ out) {
  extern __shared__ char smem[];
  __bf16* qk = (__bf16*)smem;                 // 64 x 768
  __bf16* vT = (__bf16*)(smem + 98304);       // 384 x 64
  __bf16* xa = (__bf16*)(smem + 147456);      // 64 x 384
  __bf16* Pb = (__bf16*)(smem + 196608);      // 8 x 64 x 64
  float*  ob = (float*)smem;                  // 64 x 384 f32 (reuse qk region)

  const int tid  = threadIdx.x;
  const int wave = tid >> 5;
  const int lane = tid & 31;
  const int laneM = lane & 15;
  const int kbase = (lane >> 4) << 3;
  const int hiRow = (lane >> 4) << 3;         // +8 row offset for D frags

  const int blk = blockIdx.x;
  const int b   = blk >> 6;
  const int win = blk & 63;
  const int h0  = (win >> 3) * WS;
  const int w0  = (win & 7) * WS;
  const float scale = 0.14433756729740643f;   // 48^-0.5

  // ---- Phase 0: stage input window (64 tokens x 384 ch) as bf16 ----
  for (int idx = tid; idx < TOKS * DIM; idx += 256) {
    int tok = idx & 63, c = idx >> 6;
    float v = x[(((size_t)b * DIM + c) << 12) + ((h0 + (tok >> 3)) << 6) + w0 + (tok & 7)];
    xa[tok * DIM + c] = (__bf16)v;
  }
  __syncthreads();

  // ---- Phase 1: QKV GEMM (64x384 @ 384x1152) ----
  // wave -> row-tile tm = wave&3, column tiles [tn0, tn0+36)
  {
    const int tm  = wave & 3;
    const int tn0 = (wave >> 2) * 36;
    const int rbase = tm * 16 + hiRow;
    const __bf16* arow = xa + (tm * 16 + laneM) * DIM;
    v16bf afr[KSTEPS];
#pragma unroll
    for (int kk = 0; kk < KSTEPS; ++kk) afr[kk] = ld_fragA(arow, kbase, kk * 32);

    for (int tn = tn0; tn < tn0 + 36; ++tn) {
      const __bf16* bcol = WqkvT + (size_t)(tn * 16 + laneM) * DIM;
      v8f acc = {};
#pragma unroll
      for (int kk = 0; kk < KSTEPS; ++kk) {
        v16bf bf = ld_fragB(bcol, lane, kk * 32);
        acc = __builtin_amdgcn_wmma_f32_16x16x32_bf16(false, afr[kk], false, bf,
                                                      (short)0, acc, false, false);
      }
      const int col  = tn * 16 + laneM;
      const float bias = b_qkv[col];
      if (tn < 24) {                       // Q, pre-scaled (uniform branch)
#pragma unroll
        for (int r = 0; r < 8; ++r)
          qk[(rbase + r) * 768 + col] = (__bf16)((acc[r] + bias) * scale);
      } else if (tn < 48) {                // K
#pragma unroll
        for (int r = 0; r < 8; ++r)
          qk[(rbase + r) * 768 + col] = (__bf16)(acc[r] + bias);
      } else {                             // V, transposed, packed b128 store
        union { v8bf v; __bf16 e[8]; } pk;
#pragma unroll
        for (int r = 0; r < 8; ++r) pk.e[r] = (__bf16)(acc[r] + bias);
        *(v8bf*)(vT + (col - 2 * DIM) * TOKS + rbase) = pk.v;
      }
    }
  }
  __syncthreads();

  // ---- Phase 2: attention, one head per wave ----
  const int h = wave;
  v8f s[4][4] = {};
  // S = Q Kt, K-dim = 48 padded to 64 (upper groups are exactly zero-masked)
#pragma unroll
  for (int kk = 0; kk < 2; ++kk) {
    v16bf af[4], bf[4];
#pragma unroll
    for (int tm = 0; tm < 4; ++tm) {
      const __bf16* row = qk + (tm * 16 + laneM) * 768 + h * HD;
      af[tm] = (kk == 0) ? ld_fragA(row, kbase, 0) : ld_fragA_lo(row, kbase, 32);
    }
#pragma unroll
    for (int tn = 0; tn < 4; ++tn) {
      const __bf16* col = qk + (tn * 16 + laneM) * 768 + DIM + h * HD;
      if (kk == 0) {
        bf[tn] = ld_fragB(col, lane, 0);
      } else {
        // lanes 0..15 cover k=32..47 (valid), lanes 16..31 would be k=48..63 -> zero
        v8bf lo = *(const v8bf*)(col + 32);
        v8bf hi = *(const v8bf*)(col + 40);
        bool valid = lane < 16;
        union { v16bf v; v8bf g[2]; } u;
        u.g[0] = valid ? lo : zero8();
        u.g[1] = valid ? hi : zero8();
        bf[tn] = u.v;
      }
    }
#pragma unroll
    for (int tm = 0; tm < 4; ++tm)
#pragma unroll
      for (int tn = 0; tn < 4; ++tn)
        s[tm][tn] = __builtin_amdgcn_wmma_f32_16x16x32_bf16(false, af[tm], false, bf[tn],
                                                            (short)0, s[tm][tn], false, false);
  }

  // In-register softmax over each row (cols live across lanes 0..15 of each half)
#pragma unroll
  for (int tm = 0; tm < 4; ++tm) {
#pragma unroll
    for (int r = 0; r < 8; ++r) {
      float m = s[tm][0][r];
#pragma unroll
      for (int tn = 1; tn < 4; ++tn) m = fmaxf(m, s[tm][tn][r]);
#pragma unroll
      for (int off = 1; off < 16; off <<= 1) m = fmaxf(m, __shfl_xor(m, off, 32));
      float sum = 0.f;
#pragma unroll
      for (int tn = 0; tn < 4; ++tn) {
        float p = __expf(s[tm][tn][r] - m);
        s[tm][tn][r] = p;
        sum += p;
      }
#pragma unroll
      for (int off = 1; off < 16; off <<= 1) sum += __shfl_xor(sum, off, 32);
      float inv = 1.f / sum;
#pragma unroll
      for (int tn = 0; tn < 4; ++tn) s[tm][tn][r] *= inv;
    }
  }

  // Store P (bf16) for A-fragment reload
  __bf16* Ph = Pb + h * TOKS * TOKS;
#pragma unroll
  for (int tm = 0; tm < 4; ++tm)
#pragma unroll
    for (int tn = 0; tn < 4; ++tn)
#pragma unroll
      for (int r = 0; r < 8; ++r)
        Ph[(tm * 16 + hiRow + r) * TOKS + tn * 16 + laneM] = (__bf16)s[tm][tn][r];

  // O = P @ V_h  (64x64 @ 64x48)
  v8f o[4][3] = {};
#pragma unroll
  for (int kk = 0; kk < 2; ++kk) {
    v16bf af[4], bf[3];
#pragma unroll
    for (int tm = 0; tm < 4; ++tm)
      af[tm] = ld_fragA(Ph + (tm * 16 + laneM) * TOKS, kbase, kk * 32);
#pragma unroll
    for (int tn = 0; tn < 3; ++tn)
      bf[tn] = ld_fragB(vT + (h * HD + tn * 16 + laneM) * TOKS, lane, kk * 32);
#pragma unroll
    for (int tm = 0; tm < 4; ++tm)
#pragma unroll
      for (int tn = 0; tn < 3; ++tn)
        o[tm][tn] = __builtin_amdgcn_wmma_f32_16x16x32_bf16(false, af[tm], false, bf[tn],
                                                            (short)0, o[tm][tn], false, false);
  }
  // attention output (bf16) into xa (input staging no longer needed)
#pragma unroll
  for (int tm = 0; tm < 4; ++tm)
#pragma unroll
    for (int tn = 0; tn < 3; ++tn)
#pragma unroll
      for (int r = 0; r < 8; ++r)
        xa[(tm * 16 + hiRow + r) * DIM + h * HD + tn * 16 + laneM] = (__bf16)o[tm][tn][r];
  __syncthreads();

  // ---- Phase 3: output projection (64x384 @ 384x384), f32 result in LDS ----
  // wave -> row-tile tm = wave&3, column tiles [tn0, tn0+12)
  {
    const int tm  = wave & 3;
    const int tn0 = (wave >> 2) * 12;
    const int rbase = tm * 16 + hiRow;
    const __bf16* arow = xa + (tm * 16 + laneM) * DIM;
    v16bf afr[KSTEPS];
#pragma unroll
    for (int kk = 0; kk < KSTEPS; ++kk) afr[kk] = ld_fragA(arow, kbase, kk * 32);

    for (int tn = tn0; tn < tn0 + 12; ++tn) {
      const __bf16* bcol = WprojT + (size_t)(tn * 16 + laneM) * DIM;
      v8f acc = {};
#pragma unroll
      for (int kk = 0; kk < KSTEPS; ++kk) {
        v16bf bf = ld_fragB(bcol, lane, kk * 32);
        acc = __builtin_amdgcn_wmma_f32_16x16x32_bf16(false, afr[kk], false, bf,
                                                      (short)0, acc, false, false);
      }
      const int col = tn * 16 + laneM;
#pragma unroll
      for (int r = 0; r < 8; ++r)
        ob[(rbase + r) * DIM + col] = acc[r];
    }
  }
  __syncthreads();

  // ---- Phase 4: coalesced NCHW writeback with bias ----
  for (int idx = tid; idx < TOKS * DIM; idx += 256) {
    int tok = idx & 63, c = idx >> 6;
    out[(((size_t)b * DIM + c) << 12) + ((h0 + (tok >> 3)) << 6) + w0 + (tok & 7)]
        = ob[tok * DIM + c] + b_proj[c];
  }
}

extern "C" void kernel_launch(void* const* d_in, const int* in_sizes, int n_in,
                              void* d_out, int out_size, void* d_ws, size_t ws_size,
                              hipStream_t stream) {
  const float* x      = (const float*)d_in[0];
  const float* W_qkv  = (const float*)d_in[1];
  const float* b_qkv  = (const float*)d_in[2];
  const float* W_proj = (const float*)d_in[3];
  const float* b_proj = (const float*)d_in[4];
  float* out = (float*)d_out;

  __bf16* WqkvT  = (__bf16*)d_ws;                                   // 1152*384 bf16
  __bf16* WprojT = (__bf16*)((char*)d_ws + (size_t)QKVW * DIM * 2); // 384*384 bf16

  (void)hipFuncSetAttribute((const void*)window_attention_kernel,
                            hipFuncAttributeMaxDynamicSharedMemorySize, LDS_BYTES);

  int totalW = QKVW * DIM + DIM * DIM;
  convert_weights_kernel<<<(totalW + 255) / 256, 256, 0, stream>>>(W_qkv, W_proj, WqkvT, WprojT);

  // 16 batches * 64 windows = 1024 workgroups
  window_attention_kernel<<<dim3(1024), dim3(256), LDS_BYTES, stream>>>(
      x, b_qkv, b_proj, WqkvT, WprojT, out);
}
